// DGCGRUCell_20933670601192
// MI455X (gfx1250) — matrix operand order repaired and checked by reading, not compiled, not run on hardware
//
#include <hip/hip_runtime.h>
#include <hip/hip_bf16.h>
#include <stdint.h>

// ---------------- problem dims ----------------
#define BATCH 1024
#define NJ    128
#define DIN   256
#define DOUT  512
#define DCAT  (DIN + DOUT)   // 768
#define MROWS (BATCH * NJ)   // 131072

typedef __attribute__((ext_vector_type(16))) __bf16 v16bf;
typedef __attribute__((ext_vector_type(8)))  float  v8f;

union ABFrag { v16bf v; uint4 q[2]; };

__device__ __forceinline__ uint16_t f2bfu(float f) {
    union { __bf16 b; uint16_t u; } c;
    c.b = (__bf16)f;
    return c.u;
}

#define WMMA_BF16(a, b, c) \
    __builtin_amdgcn_wmma_f32_16x16x32_bf16(false, (a), false, (b), (short)0, (c), false, false)

// A fragment: row-major A, 16x32 tile row = one lane's M. lane<16 holds K [0..7],[16..23];
// lane>=16 holds K [8..15],[24..31]  (per CDNA5 ISA 16-bit A layout).
__device__ __forceinline__ v16bf ldA(const uint16_t* rowPtr, int kk, int lane) {
    ABFrag f;
    const uint16_t* p = rowPtr + kk + ((lane >> 4) << 3);
    f.q[0] = *(const uint4*)(p);
    f.q[1] = *(const uint4*)(p + 16);
    return f.v;
}

// B fragment from B^T (row-major [N][K]): lane<16 holds K [0..7],[16..23], lane>=16 K [8..15],[24..31],
// column N = tn + (lane&15). Two 16-byte loads per lane.
__device__ __forceinline__ v16bf ldB(const uint16_t* colPtr, int kk, int lane) {
    ABFrag f;
    const uint16_t* p = colPtr + kk + ((lane >> 4) << 3);
    f.q[0] = *(const uint4*)(p);
    f.q[1] = *(const uint4*)(p + 16);
    return f.v;
}

// ---------------- prep kernels ----------------

__global__ void k_cvt_weights(const float* __restrict__ Wz, const float* __restrict__ Wr,
                              const float* __restrict__ Wh, const float* __restrict__ Wn,
                              uint16_t* __restrict__ Wzb, uint16_t* __restrict__ Wrb,
                              uint16_t* __restrict__ Whb, uint16_t* __restrict__ Wnb) {
    int i = blockIdx.x * blockDim.x + threadIdx.x;
    const int NW = DOUT * DCAT;
    if (i < NW) {
        Wzb[i] = f2bfu(Wz[i]);
        Wrb[i] = f2bfu(Wr[i]);
        Whb[i] = f2bfu(Wh[i]);
    }
    if (i < NJ * NJ) Wnb[i] = f2bfu(Wn[i]);
}

// Per batch: L = (A+I) / (D_i*D_j + 1e-12), D = sqrt(rowsum(A+I)), stored bf16 row-major.
__global__ void k_prep_L(const float* __restrict__ A, uint16_t* __restrict__ Lbf) {
    __shared__ float Ds[NJ];
    int b = blockIdx.x;
    int i = threadIdx.x;                 // 128 threads
    const float* Ab = A + (size_t)b * NJ * NJ;
    float s = 1.0f;                      // self loop
    for (int j = 0; j < NJ; ++j) s += Ab[i * NJ + j];
    Ds[i] = sqrtf(s);
    __syncthreads();
    float Di = Ds[i];
    uint16_t* Lo = Lbf + (size_t)b * NJ * NJ + (size_t)i * NJ;
    for (int j = 0; j < NJ; ++j) {
        float v = Ab[i * NJ + j] + (i == j ? 1.0f : 0.0f);
        Lo[j] = f2bfu(v / (Di * Ds[j] + 1e-12f));
    }
}

// XcT[b][d][j] = concat(x,h)[b][j][d] (bf16, B-operand layout for L@X),
// and XH[b*NJ+j][d] = x for d < DIN (A-operand layout for the H GEMM).
__global__ void k_prep_X(const float* __restrict__ x, const float* __restrict__ h,
                         uint16_t* __restrict__ XcT, uint16_t* __restrict__ XH) {
    long long idx = (long long)blockIdx.x * blockDim.x + threadIdx.x;
    const long long total = (long long)BATCH * DCAT * NJ;
    if (idx >= total) return;
    int b = (int)(idx / (DCAT * NJ));
    int rem = (int)(idx % (DCAT * NJ));
    int d = rem / NJ;
    int j = rem % NJ;
    float v = (d < DIN) ? x[((size_t)b * NJ + j) * DIN + d]
                        : h[((size_t)b * NJ + j) * DOUT + (d - DIN)];
    XcT[idx] = f2bfu(v);
    if (d < DIN) XH[((size_t)b * NJ + j) * DCAT + d] = f2bfu(v);
}

// ---------------- WMMA GEMM kernels ----------------

// LXT[b][d][m] = (L @ Xc)^T : A = L [128x128], B^T = XcT [768x128].
// 2x2 register blocking: wave computes 32x32; transposed bf16 stores (16B each).
__global__ void k_gemm_LX(const uint16_t* __restrict__ Lbf, const uint16_t* __restrict__ XcT,
                          uint16_t* __restrict__ LXT) {
    int b = blockIdx.x;
    int lane = threadIdx.x & 31;
    int wid = threadIdx.x >> 5;
    int tile = blockIdx.y * 8 + wid;       // 96 tiles: 4 (M32) x 24 (N32)
    int tm = (tile & 3) << 5;
    int tn = (tile >> 2) << 5;
    const uint16_t* Ar0 = Lbf + (size_t)b * NJ * NJ + (size_t)(tm + (lane & 15)) * NJ;
    const uint16_t* Ar1 = Ar0 + (size_t)16 * NJ;
    const uint16_t* Bc0 = XcT + (size_t)b * DCAT * NJ + (size_t)(tn + (lane & 15)) * NJ;
    const uint16_t* Bc1 = Bc0 + (size_t)16 * NJ;
    v8f a00 = {}, a01 = {}, a10 = {}, a11 = {};
#pragma unroll
    for (int kk = 0; kk < NJ; kk += 32) {
        v16bf fa0 = ldA(Ar0, kk, lane);
        v16bf fa1 = ldA(Ar1, kk, lane);
        v16bf fb0 = ldB(Bc0, kk, lane);
        v16bf fb1 = ldB(Bc1, kk, lane);
        a00 = WMMA_BF16(fa0, fb0, a00);
        a01 = WMMA_BF16(fa0, fb1, a01);
        a10 = WMMA_BF16(fa1, fb0, a10);
        a11 = WMMA_BF16(fa1, fb1, a11);
    }
    uint16_t* Ob = LXT + (size_t)b * DCAT * NJ;
    int half8 = (lane >> 4) << 3;
    int nl = lane & 15;
#pragma unroll
    for (int mi = 0; mi < 2; ++mi) {
#pragma unroll
        for (int ni = 0; ni < 2; ++ni) {
            v8f acc = mi == 0 ? (ni == 0 ? a00 : a01) : (ni == 0 ? a10 : a11);
            int n = tn + ni * 16 + nl;
            int m0 = tm + mi * 16 + half8;
            union { uint16_t s[8]; uint4 q; } ov;
#pragma unroll
            for (int r = 0; r < 8; ++r) ov.s[r] = f2bfu(acc[r]);
            *(uint4*)(Ob + (size_t)n * NJ + m0) = ov.q;
        }
    }
}

// Wt[b][i][k] = relu(L @ Wn^T + bn), row-major bf16. B^T = Wn row-major [128x128].
__global__ void k_gemm_W(const uint16_t* __restrict__ Lbf, const uint16_t* __restrict__ Wnb,
                         const float* __restrict__ bn, uint16_t* __restrict__ Wt) {
    int b = blockIdx.x;
    int lane = threadIdx.x & 31;
    int wid = threadIdx.x >> 5;
    int tile = blockIdx.y * 8 + wid;       // 64 tiles: 8 x 8
    int tm = (tile & 7) << 4;
    int tn = (tile >> 3) << 4;
    const uint16_t* Ar = Lbf + (size_t)b * NJ * NJ + (size_t)(tm + (lane & 15)) * NJ;
    const uint16_t* Bc = Wnb + (size_t)(tn + (lane & 15)) * NJ;
    v8f acc = {};
#pragma unroll
    for (int kk = 0; kk < NJ; kk += 32) {
        v16bf a = ldA(Ar, kk, lane);
        v16bf bm = ldB(Bc, kk, lane);
        acc = WMMA_BF16(a, bm, acc);
    }
    int n = tn + (lane & 15);
    int m0 = tm + ((lane >> 4) << 3);
    float bias = bn[n];
    uint16_t* O = Wt + (size_t)b * NJ * NJ;
#pragma unroll
    for (int r = 0; r < 8; ++r) {
        float v = acc[r] + bias;
        O[(size_t)(m0 + r) * NJ + n] = f2bfu(v > 0.0f ? v : 0.0f);
    }
}

// Y[b*NJ+i][d] = W @ LX, row-major bf16. A = Wt [128x128], B^T = LXT [768x128]. 2x2 blocking.
__global__ void k_gemm_Y(const uint16_t* __restrict__ Wt, const uint16_t* __restrict__ LXT,
                         uint16_t* __restrict__ Y) {
    int b = blockIdx.x;
    int lane = threadIdx.x & 31;
    int wid = threadIdx.x >> 5;
    int tile = blockIdx.y * 8 + wid;       // 96 tiles: 4 (M32) x 24 (N32)
    int tm = (tile & 3) << 5;
    int tn = (tile >> 2) << 5;
    const uint16_t* Ar0 = Wt + (size_t)b * NJ * NJ + (size_t)(tm + (lane & 15)) * NJ;
    const uint16_t* Ar1 = Ar0 + (size_t)16 * NJ;
    const uint16_t* Bc0 = LXT + (size_t)b * DCAT * NJ + (size_t)(tn + (lane & 15)) * NJ;
    const uint16_t* Bc1 = Bc0 + (size_t)16 * NJ;
    v8f a00 = {}, a01 = {}, a10 = {}, a11 = {};
#pragma unroll
    for (int kk = 0; kk < NJ; kk += 32) {
        v16bf fa0 = ldA(Ar0, kk, lane);
        v16bf fa1 = ldA(Ar1, kk, lane);
        v16bf fb0 = ldB(Bc0, kk, lane);
        v16bf fb1 = ldB(Bc1, kk, lane);
        a00 = WMMA_BF16(fa0, fb0, a00);
        a01 = WMMA_BF16(fa0, fb1, a01);
        a10 = WMMA_BF16(fa1, fb0, a10);
        a11 = WMMA_BF16(fa1, fb1, a11);
    }
    uint16_t* O = Y + (size_t)b * NJ * DCAT;
    int half8 = (lane >> 4) << 3;
    int nl = lane & 15;
#pragma unroll
    for (int mi = 0; mi < 2; ++mi) {
#pragma unroll
        for (int ni = 0; ni < 2; ++ni) {
            v8f acc = mi == 0 ? (ni == 0 ? a00 : a01) : (ni == 0 ? a10 : a11);
            int n = tn + ni * 16 + nl;
            int m0 = tm + mi * 16 + half8;
#pragma unroll
            for (int r = 0; r < 8; ++r) O[(size_t)(m0 + r) * DCAT + n] = f2bfu(acc[r]);
        }
    }
}

// Fused Z/R gate GEMM over [131072 x 768] x [768 x 512]. Wave computes 32 rows x 16 cols
// for BOTH gates: A0/A1 shared across Z and R, Bz/Br shared across both M sub-tiles
// -> 4 WMMA per 4 fragment loads. Epilogue: Zbuf = sigmoid(.+bz) (f32),
// XH[:,DIN+o] = h * sigmoid(.+br) (bf16).
__global__ void k_gemm_gates(const uint16_t* __restrict__ Y,
                             const uint16_t* __restrict__ Wzb, const uint16_t* __restrict__ Wrb,
                             const float* __restrict__ bz, const float* __restrict__ br,
                             const float* __restrict__ h,
                             float* __restrict__ Zbuf, uint16_t* __restrict__ XH) {
    int lane = threadIdx.x & 31;
    int wid = threadIdx.x >> 5;
    long long w = (long long)blockIdx.x * 8 + wid;  // 131072 waves: 4096 (M32) x 32 (N16)
    int tm = (int)(w & 4095) << 5;
    int tn = (int)(w >> 12) << 4;
    const uint16_t* Ar0 = Y + (size_t)(tm + (lane & 15)) * DCAT;
    const uint16_t* Ar1 = Ar0 + (size_t)16 * DCAT;
    const uint16_t* Bz = Wzb + (size_t)(tn + (lane & 15)) * DCAT;
    const uint16_t* Br = Wrb + (size_t)(tn + (lane & 15)) * DCAT;
    v8f aZ0 = {}, aZ1 = {}, aR0 = {}, aR1 = {};
#pragma unroll 4
    for (int kk = 0; kk < DCAT; kk += 32) {
        v16bf fa0 = ldA(Ar0, kk, lane);
        v16bf fa1 = ldA(Ar1, kk, lane);
        v16bf fbz = ldB(Bz, kk, lane);
        v16bf fbr = ldB(Br, kk, lane);
        aZ0 = WMMA_BF16(fa0, fbz, aZ0);
        aZ1 = WMMA_BF16(fa1, fbz, aZ1);
        aR0 = WMMA_BF16(fa0, fbr, aR0);
        aR1 = WMMA_BF16(fa1, fbr, aR1);
    }
    int col = tn + (lane & 15);
    int half8 = (lane >> 4) << 3;
    float bzv = bz[col], brv = br[col];
#pragma unroll
    for (int mi = 0; mi < 2; ++mi) {
        v8f accZ = mi == 0 ? aZ0 : aZ1;
        v8f accR = mi == 0 ? aR0 : aR1;
        int m0 = tm + mi * 16 + half8;
#pragma unroll
        for (int r = 0; r < 8; ++r) {
            int row = m0 + r;
            float z = 1.0f / (1.0f + __expf(-(accZ[r] + bzv)));
            float rg = 1.0f / (1.0f + __expf(-(accR[r] + brv)));
            Zbuf[(size_t)row * DOUT + col] = z;
            float hv = h[(size_t)row * DOUT + col];
            XH[(size_t)row * DCAT + DIN + col] = f2bfu(hv * rg);
        }
    }
}

// H GEMM + final combine: out = Z*h + (1-Z)*tanh(XH @ Wh^T + bh). 2x2 blocking (32x32/wave).
__global__ void k_gemm_out(const uint16_t* __restrict__ XH, const uint16_t* __restrict__ Whb,
                           const float* __restrict__ bh, const float* __restrict__ Zbuf,
                           const float* __restrict__ h, float* __restrict__ out) {
    int lane = threadIdx.x & 31;
    int wid = threadIdx.x >> 5;
    long long w = (long long)blockIdx.x * 8 + wid;  // 65536 waves: 4096 (M32) x 16 (N32)
    int tm = (int)(w & 4095) << 5;
    int tn = (int)(w >> 12) << 5;
    const uint16_t* Ar0 = XH + (size_t)(tm + (lane & 15)) * DCAT;
    const uint16_t* Ar1 = Ar0 + (size_t)16 * DCAT;
    const uint16_t* Bc0 = Whb + (size_t)(tn + (lane & 15)) * DCAT;
    const uint16_t* Bc1 = Bc0 + (size_t)16 * DCAT;
    v8f a00 = {}, a01 = {}, a10 = {}, a11 = {};
#pragma unroll 4
    for (int kk = 0; kk < DCAT; kk += 32) {
        v16bf fa0 = ldA(Ar0, kk, lane);
        v16bf fa1 = ldA(Ar1, kk, lane);
        v16bf fb0 = ldB(Bc0, kk, lane);
        v16bf fb1 = ldB(Bc1, kk, lane);
        a00 = WMMA_BF16(fa0, fb0, a00);
        a01 = WMMA_BF16(fa0, fb1, a01);
        a10 = WMMA_BF16(fa1, fb0, a10);
        a11 = WMMA_BF16(fa1, fb1, a11);
    }
    int half8 = (lane >> 4) << 3;
    int nl = lane & 15;
#pragma unroll
    for (int mi = 0; mi < 2; ++mi) {
#pragma unroll
        for (int ni = 0; ni < 2; ++ni) {
            v8f acc = mi == 0 ? (ni == 0 ? a00 : a01) : (ni == 0 ? a10 : a11);
            int col = tn + ni * 16 + nl;
            int m0 = tm + mi * 16 + half8;
            float bias = bh[col];
#pragma unroll
            for (int r = 0; r < 8; ++r) {
                int row = m0 + r;
                float Hv = tanhf(acc[r] + bias);
                float z = Zbuf[(size_t)row * DOUT + col];
                float hv = h[(size_t)row * DOUT + col];
                out[(size_t)row * DOUT + col] = z * hv + (1.0f - z) * Hv;
            }
        }
    }
}

// ---------------- host launch ----------------

extern "C" void kernel_launch(void* const* d_in, const int* in_sizes, int n_in,
                              void* d_out, int out_size, void* d_ws, size_t ws_size,
                              hipStream_t stream) {
    const float* x  = (const float*)d_in[0];
    const float* h  = (const float*)d_in[1];
    const float* A  = (const float*)d_in[2];
    const float* Wz = (const float*)d_in[3];
    const float* bz = (const float*)d_in[4];
    const float* Wr = (const float*)d_in[5];
    const float* br = (const float*)d_in[6];
    const float* Wh = (const float*)d_in[7];
    const float* bh = (const float*)d_in[8];
    const float* Wn = (const float*)d_in[9];
    const float* bn = (const float*)d_in[10];
    float* out = (float*)d_out;

    char* ws = (char*)d_ws;
    size_t off = 0;
    auto carve = [&](size_t bytes) -> char* {
        char* p = ws + off;
        off = (off + bytes + 255) & ~(size_t)255;
        return p;
    };
    uint16_t* Wzb = (uint16_t*)carve((size_t)DOUT * DCAT * 2);
    uint16_t* Wrb = (uint16_t*)carve((size_t)DOUT * DCAT * 2);
    uint16_t* Whb = (uint16_t*)carve((size_t)DOUT * DCAT * 2);
    uint16_t* Wnb = (uint16_t*)carve((size_t)NJ * NJ * 2);
    uint16_t* Lbf = (uint16_t*)carve((size_t)BATCH * NJ * NJ * 2);
    uint16_t* XcT = (uint16_t*)carve((size_t)BATCH * DCAT * NJ * 2);
    uint16_t* XH  = (uint16_t*)carve((size_t)MROWS * DCAT * 2);
    uint16_t* LXT = (uint16_t*)carve((size_t)BATCH * DCAT * NJ * 2);
    uint16_t* Wt  = (uint16_t*)carve((size_t)BATCH * NJ * NJ * 2);
    uint16_t* Y   = (uint16_t*)carve((size_t)MROWS * DCAT * 2);
    float*    Zb  = (float*)carve((size_t)MROWS * DOUT * 4);

    // stage 0: conversions / graph prep
    k_cvt_weights<<<(DOUT * DCAT + 255) / 256, 256, 0, stream>>>(Wz, Wr, Wh, Wn, Wzb, Wrb, Whb, Wnb);
    k_prep_L<<<BATCH, NJ, 0, stream>>>(A, Lbf);
    {
        long long total = (long long)BATCH * DCAT * NJ;
        k_prep_X<<<(unsigned)((total + 255) / 256), 256, 0, stream>>>(x, h, XcT, XH);
    }
    // stage 1: LX^T = (L @ X)^T   (96 waves/batch, 12 blocks of 8 waves)
    k_gemm_LX<<<dim3(BATCH, 12), 256, 0, stream>>>(Lbf, XcT, LXT);
    // stage 2: W = relu(L @ Wn^T + bn)
    k_gemm_W<<<dim3(BATCH, 8), 256, 0, stream>>>(Lbf, Wnb, bn, Wt);
    // stage 3: Y = W @ LX
    k_gemm_Y<<<dim3(BATCH, 12), 256, 0, stream>>>(Wt, LXT, Y);
    // stage 4: fused Z/R gates -> Zbuf, XH[:,256:]   (131072 waves / 8 = 16384 blocks)
    k_gemm_gates<<<16384, 256, 0, stream>>>(Y, Wzb, Wrb, bz, br, h, Zb, XH);
    // stage 5: H GEMM + final combine                 (65536 waves / 8 = 8192 blocks)
    k_gemm_out<<<8192, 256, 0, stream>>>(XH, Whb, bh, Zb, h, out);
}
